// PSMMixSeqEmbedding_40596030881949
// MI455X (gfx1250) — matrix operand order repaired and checked by reading, not compiled, not run on hardware
//
#include <hip/hip_runtime.h>
#include <hip/hip_bf16.h>

typedef __attribute__((ext_vector_type(16))) __bf16 v16bf;
typedef __attribute__((ext_vector_type(8)))  __bf16 v8bf;
typedef __attribute__((ext_vector_type(8)))  float  v8f;

__device__ __forceinline__ __bf16 f2bf(float x){ return (__bf16)x; }

__device__ __forceinline__ v8f wmma_bf16(v16bf a, v16bf b, v8f c){
  // D = A(16x32 bf16) * B(32x16 bf16) + C(16x16 f32)
  return __builtin_amdgcn_wmma_f32_16x16x32_bf16(false, a, false, b, (short)0, c, false, false);
}

__device__ __forceinline__ v16bf combine16(v8bf lo, v8bf hi){
  v16bf r;
#pragma unroll
  for (int i = 0; i < 8; i++){ r[i] = lo[i]; r[8 + i] = hi[i]; }
  return r;
}

// ---------------------------------------------------------------------------
// Weight prep: transpose + f32->bf16 for all WMMA B-operands.
// t_w1t/t_w2t: [1024n][1024k], bp_w1t: [512n][128k], bp_w2t: [32n][512k]
// ---------------------------------------------------------------------------
__global__ void prep_weights(const float* __restrict__ tw1, const float* __restrict__ tw2,
                             const float* __restrict__ bw1, const float* __restrict__ bw2,
                             __bf16* __restrict__ tw1t, __bf16* __restrict__ tw2t,
                             __bf16* __restrict__ bw1t, __bf16* __restrict__ bw2t){
  int idx = blockIdx.x * blockDim.x + threadIdx.x;
  if (idx < 1048576){
    int n = idx >> 10, k = idx & 1023;
    tw1t[idx] = f2bf(tw1[k * 1024 + n]);
  } else if (idx < 2097152){
    int j = idx - 1048576; int n = j >> 10, k = j & 1023;
    tw2t[j] = f2bf(tw2[k * 1024 + n]);
  } else if (idx < 2162688){
    int j = idx - 2097152; int n = j >> 7, k = j & 127;
    bw1t[j] = f2bf(bw1[k * 512 + n]);
  } else if (idx < 2179072){
    int j = idx - 2162688; int n = j >> 9, k = j & 511;
    bw2t[j] = f2bf(bw2[k * 32 + n]);
  }
}

// ---------------------------------------------------------------------------
// x = embed_w[tok] + (mol ? sum_f atom_feat_w[attr_f] : 0) + chain_w[chain]
// Also writes padding_mask. One block per (b,l) row, float4 over D=1024.
// ---------------------------------------------------------------------------
__global__ __launch_bounds__(256) void embed_kernel(
    const int* __restrict__ token_id, const int* __restrict__ chain_ids,
    const int* __restrict__ is_periodic, const int* __restrict__ node_attr,
    const float* __restrict__ embed_w, const float* __restrict__ atom_feat_w,
    const float* __restrict__ chain_w,
    float* __restrict__ out_x0, float* __restrict__ out_x4, float* __restrict__ out_pad){
  int row = blockIdx.x;                 // [0, B*L)
  int b = row >> 9;
  int tok = token_id[row];
  int chain = chain_ids[row];
  bool mol = (tok <= 129) && (tok > 1) && (is_periodic[b] == 0);
  int t = threadIdx.x;
  int d = t * 4;
  float4 v = *(const float4*)(embed_w + (long)tok * 1024 + d);
  float4 c = *(const float4*)(chain_w + (long)chain * 1024 + d);
  v.x += c.x; v.y += c.y; v.z += c.z; v.w += c.w;
  if (mol){
#pragma unroll
    for (int f = 1; f < 9; f++){
      int a = node_attr[row * 9 + f];
      float4 w = *(const float4*)(atom_feat_w + (long)a * 1024 + d);
      v.x += w.x; v.y += w.y; v.z += w.z; v.w += w.w;
    }
  }
  *(float4*)(out_x0 + (long)row * 1024 + d) = v;
  *(float4*)(out_x4 + (long)row * 1024 + d) = v;
  if (t == 0) out_pad[row] = (tok == 0) ? 1.0f : 0.0f;
}

// ---------------------------------------------------------------------------
// Sinusoidal embedding, stored bf16 for WMMA A-operand.
// ---------------------------------------------------------------------------
__global__ void time_sinus(const int* __restrict__ time_step, __bf16* __restrict__ te){
  int idx = blockIdx.x * blockDim.x + threadIdx.x;  // [0, 1024*1024)
  int row = idx >> 10;
  int d = idx & 1023;
  float t = (float)time_step[row];
  int h = (d < 512) ? d : (d - 512);
  float f = __expf(-9.2103403719761836f * ((float)h * (1.0f / 512.0f)));
  float a = t * f;
  te[idx] = f2bf((d < 512) ? __sinf(a) : __cosf(a));
}

// ---------------------------------------------------------------------------
// Generic 1024x1024x1024 bf16 WMMA GEMM for the time-step MLP.
// mode 0: dst_bf = bf16(silu(A@Bt^T + bias))      (stage 1)
// mode 1: dst_f  = clean ? 0 : (A@Bt^T + bias)    (stage 2)
// One 16x16 tile per wave, 8 waves/block, K-loop of 32 (K=1024).
// ---------------------------------------------------------------------------
__global__ __launch_bounds__(256) void time_gemm(
    const __bf16* __restrict__ A, const __bf16* __restrict__ Bt,
    const float* __restrict__ bias, __bf16* __restrict__ dst_bf,
    float* __restrict__ dst_f, const int* __restrict__ clean, int mode){
  int wave = threadIdx.x >> 5;
  int lane = threadIdx.x & 31;
  int half = lane >> 4;
  int idx  = lane & 15;
  int tile = blockIdx.x * 8 + wave;
  int rt = tile >> 6;       // row tile (64 tiles)
  int ct = tile & 63;       // col tile
  v8f acc = {};
  for (int s = 0; s < 32; s++){
    const v8bf* ap = (const v8bf*)(A + ((long)(rt * 16 + idx)) * 1024 + s * 32 + half * 8);
    v16bf af = combine16(ap[0], ap[2]);     // K: kb+half*8..+7 and kb+16+half*8..+7
    const v8bf* bp = (const v8bf*)(Bt + ((long)(ct * 16 + idx)) * 1024 + s * 32 + half * 8);
    v16bf bf = combine16(bp[0], bp[2]);     // K: kb+half*8..+7 and kb+16+half*8..+7
    acc = wmma_bf16(af, bf, acc);
  }
#pragma unroll
  for (int r = 0; r < 8; r++){
    int row = rt * 16 + half * 8 + r;
    int col = ct * 16 + idx;
    float v = acc[r] + bias[col];
    if (mode == 0){
      float s = v * (1.0f / (1.0f + __expf(-v)));
      dst_bf[(long)row * 1024 + col] = f2bf(s);
    } else {
      dst_f[(long)row * 1024 + col] = (clean[row] != 0) ? 0.0f : v;
    }
  }
}

// ---------------------------------------------------------------------------
// Fused bond-bias pipeline: gather e -> GEMM1(128->512) -> LN -> ReLU ->
// GEMM2(512->32) -> mask -> transposed scatter + mean-over-heads.
// 4 waves/block, 16 rows/wave, A-fragments live in registers, intermediate
// h staged as bf16 in wave-private LDS (64KB/block).
// ---------------------------------------------------------------------------
__global__ __launch_bounds__(128) void bond_bias(
    const int* __restrict__ token_id, const int* __restrict__ is_periodic,
    const int* __restrict__ adj, const int* __restrict__ node_type_edge,
    const float* __restrict__ bond_w, const __bf16* __restrict__ w1t,
    const __bf16* __restrict__ w2t, const float* __restrict__ lng,
    const float* __restrict__ lnb, float* __restrict__ out_gab,
    float* __restrict__ mean_ws){
  __shared__ __attribute__((aligned(16))) __bf16 c1h[4][16 * 512];
  int wave = threadIdx.x >> 5;
  int lane = threadIdx.x & 31;
  int half = lane >> 4;
  int idx  = lane & 15;
  long rowBase = ((long)blockIdx.x * 4 + wave) * 16;
  long g = rowBase + idx;                         // this lane's row
  int b = (int)(g >> 18);
  int i = (int)((g >> 9) & 511);
  int j = (int)(g & 511);
  int tok_i = token_id[(b << 9) + i];
  int tok_j = token_id[(b << 9) + j];
  bool per = is_periodic[b] != 0;
  bool mol_i = (tok_i <= 129) && (tok_i > 1) && !per;
  bool mol_j = (tok_j <= 129) && (tok_j > 1) && !per;
  int mm = (mol_i && mol_j) ? 1 : 0;
  int pmask = (mm && (adj[g] != 0)) ? 1 : 0;
  int edge = node_type_edge[g];
  const float* brow = bond_w + (long)edge * 128;
  float sc = pmask ? 1.0f : 0.0f;

  // A fragments (16 rows x 128 K) held in registers for all 32 N-tiles.
  v16bf afr[4];
#pragma unroll
  for (int s = 0; s < 4; s++){
    int base = s * 32 + half * 8;
#pragma unroll
    for (int e = 0; e < 8; e++){
      afr[s][e]     = f2bf(brow[base + e] * sc);
      afr[s][8 + e] = f2bf(brow[base + 16 + e] * sc);
    }
  }

  // GEMM1 + on-the-fly LN statistics (f32 accumulators).
  float s1[8], s2[8];
#pragma unroll
  for (int r = 0; r < 8; r++){ s1[r] = 0.0f; s2[r] = 0.0f; }
  for (int nt = 0; nt < 32; nt++){
    v8f acc = {};
#pragma unroll
    for (int s = 0; s < 4; s++){
      const v8bf* bp = (const v8bf*)(w1t + ((long)(nt * 16 + idx)) * 128 + s * 32 + half * 8);
      acc = wmma_bf16(afr[s], combine16(bp[0], bp[2]), acc);
    }
#pragma unroll
    for (int r = 0; r < 8; r++){
      float v = acc[r];
      s1[r] += v; s2[r] += v * v;
      c1h[wave][(half * 8 + r) * 512 + nt * 16 + idx] = f2bf(v);
    }
  }
  // Reduce stats across the 16 lanes of each half (rows are half-local).
#pragma unroll
  for (int r = 0; r < 8; r++){
    for (int mk = 1; mk < 16; mk <<= 1){
      s1[r] += __shfl_xor(s1[r], mk, 32);
      s2[r] += __shfl_xor(s2[r], mk, 32);
    }
  }
  __syncthreads();

  // LayerNorm + ReLU in place (bf16).
  float mean[8], rsv[8];
#pragma unroll
  for (int r = 0; r < 8; r++){
    mean[r] = s1[r] * (1.0f / 512.0f);
    float var = s2[r] * (1.0f / 512.0f) - mean[r] * mean[r];
    rsv[r] = rsqrtf(var + 1e-5f);
  }
#pragma unroll
  for (int r = 0; r < 8; r++){
    int m = half * 8 + r;
    for (int c0 = 0; c0 < 32; c0++){
      int c = idx * 32 + c0;
      float v = (float)c1h[wave][m * 512 + c];
      v = (v - mean[r]) * rsv[r] * lng[c] + lnb[c];
      c1h[wave][m * 512 + c] = f2bf(v > 0.0f ? v : 0.0f);
    }
  }
  __syncthreads();

  // GEMM2: [16x512] @ [512x32] -> two 16x16 tiles.
  v8f acc2a = {}, acc2b = {};
  for (int s = 0; s < 16; s++){
    const v8bf* hp = (const v8bf*)&c1h[wave][idx * 512 + s * 32 + half * 8];
    v16bf a2 = combine16(hp[0], hp[2]);
    const v8bf* b0 = (const v8bf*)(w2t + ((long)idx) * 512 + s * 32 + half * 8);
    const v8bf* b1 = (const v8bf*)(w2t + ((long)(16 + idx)) * 512 + s * 32 + half * 8);
    acc2a = wmma_bf16(a2, combine16(b0[0], b0[2]), acc2a);
    acc2b = wmma_bf16(a2, combine16(b1[0], b1[2]), acc2b);
  }

  // Epilogue: mask, transposed scatter to [B,H,L,L], mean over H.
#pragma unroll
  for (int r = 0; r < 8; r++){
    int m = half * 8 + r;
    long gr = rowBase + m;
    int bb = (int)(gr >> 18), ii = (int)((gr >> 9) & 511), jj = (int)(gr & 511);
    int mmr = __shfl(mm, m, 32);            // lane m holds row rowBase+m's mask
    float v0 = mmr ? acc2a[r] : 0.0f;
    float v1 = mmr ? acc2b[r] : 0.0f;
    out_gab[(((long)(bb * 32 + idx))      << 18) + (ii << 9) + jj] = v0;
    out_gab[(((long)(bb * 32 + 16 + idx)) << 18) + (ii << 9) + jj] = v1;
    float rsum = v0 + v1;
    for (int mk = 1; mk < 16; mk <<= 1) rsum += __shfl_xor(rsum, mk, 32);
    if (idx == 0) mean_ws[gr] = rsum * (1.0f / 32.0f);
  }
}

// ---------------------------------------------------------------------------
// x_p_i = silu(LN(x) @ pp_w1) @ pp_w2   (one block per (b,l) row)
// ---------------------------------------------------------------------------
__global__ __launch_bounds__(256) void pair_proj(
    const float* __restrict__ x, const float* __restrict__ g, const float* __restrict__ bt,
    const float* __restrict__ w1, const float* __restrict__ w2, float* __restrict__ xpi){
  __shared__ float xn[1024];
  __shared__ float red[256];
  __shared__ float t1[64];
  int row = blockIdx.x;
  int t = threadIdx.x;
  const float* xr = x + (long)row * 1024;
  float4 v = *(const float4*)(xr + t * 4);
  float s = v.x + v.y + v.z + v.w;
  float q = v.x * v.x + v.y * v.y + v.z * v.z + v.w * v.w;
  red[t] = s; __syncthreads();
  for (int off = 128; off > 0; off >>= 1){ if (t < off) red[t] += red[t + off]; __syncthreads(); }
  float mean = red[0] * (1.0f / 1024.0f);
  __syncthreads();
  red[t] = q; __syncthreads();
  for (int off = 128; off > 0; off >>= 1){ if (t < off) red[t] += red[t + off]; __syncthreads(); }
  float var = red[0] * (1.0f / 1024.0f) - mean * mean;
  float rs = rsqrtf(var + 1e-5f);
  __syncthreads();
#pragma unroll
  for (int u = 0; u < 4; u++){
    int d = t * 4 + u;
    xn[d] = ((&v.x)[u] - mean) * rs * g[d] + bt[d];
  }
  __syncthreads();
  int p = t & 63, qq = t >> 6;
  float acc = 0.0f;
  for (int d0 = 0; d0 < 256; d0++){
    int d = qq * 256 + d0;
    acc += xn[d] * w1[d * 64 + p];
  }
  red[t] = acc; __syncthreads();
  if (t < 64){
    float z = red[t] + red[t + 64] + red[t + 128] + red[t + 192];
    t1[t] = z * (1.0f / (1.0f + __expf(-z)));
  }
  __syncthreads();
  if (t < 64){
    float o = 0.0f;
    for (int k = 0; k < 64; k++) o += t1[k] * w2[k * 64 + t];
    xpi[(long)row * 64 + t] = o;
  }
}

// ---------------------------------------------------------------------------
// x_pair[l,k,b,p] = xpi[b,l,p]*xpi[b,k,p] + mean[b,l,k]   (pure bandwidth)
// ---------------------------------------------------------------------------
__global__ __launch_bounds__(256) void xpair_kernel(
    const float* __restrict__ xpi, const float* __restrict__ mean_ws, float* __restrict__ out){
  long idx4 = (long)blockIdx.x * blockDim.x + threadIdx.x;   // 8388608 float4s
  long e = idx4 * 4;
  int p = (int)(e & 63);
  long r = e >> 6;
  int b = (int)(r & 1);
  long lk = r >> 1;
  int k = (int)(lk & 511);
  int l = (int)(lk >> 9);
  float4 a = *(const float4*)(xpi + ((long)(b * 512 + l)) * 64 + p);
  float4 c = *(const float4*)(xpi + ((long)(b * 512 + k)) * 64 + p);
  float mn = mean_ws[((long)b << 18) + ((long)l << 9) + k];
  float4 o;
  o.x = a.x * c.x + mn; o.y = a.y * c.y + mn;
  o.z = a.z * c.z + mn; o.w = a.w * c.w + mn;
  *(float4*)(out + e) = o;
}

// ---------------------------------------------------------------------------
extern "C" void kernel_launch(void* const* d_in, const int* in_sizes, int n_in,
                              void* d_out, int out_size, void* d_ws, size_t ws_size,
                              hipStream_t stream){
  (void)in_sizes; (void)n_in; (void)out_size; (void)ws_size;
  const int*   token_id      = (const int*)d_in[0];
  const int*   chain_ids     = (const int*)d_in[1];
  const int*   is_periodic   = (const int*)d_in[2];
  const int*   time_step     = (const int*)d_in[3];
  const int*   clean_mask    = (const int*)d_in[4];
  const int*   node_attr     = (const int*)d_in[5];
  const int*   adj           = (const int*)d_in[6];
  const int*   node_type_edge= (const int*)d_in[7];
  const float* embed_w       = (const float*)d_in[8];
  const float* atom_feat_w   = (const float*)d_in[9];
  const float* chain_w       = (const float*)d_in[10];
  const float* t_w1          = (const float*)d_in[11];
  const float* t_b1          = (const float*)d_in[12];
  const float* t_w2          = (const float*)d_in[13];
  const float* t_b2          = (const float*)d_in[14];
  const float* bond_w        = (const float*)d_in[15];
  const float* bp_w1         = (const float*)d_in[16];
  const float* bp_ln_g       = (const float*)d_in[17];
  const float* bp_ln_b       = (const float*)d_in[18];
  const float* bp_w2         = (const float*)d_in[19];
  const float* pp_ln_g       = (const float*)d_in[20];
  const float* pp_ln_b       = (const float*)d_in[21];
  const float* pp_w1         = (const float*)d_in[22];
  const float* pp_w2         = (const float*)d_in[23];

  float* out     = (float*)d_out;
  float* out_x0  = out;                  // [B,L,D]   1048576
  float* out_pad = out + 1048576;        // [B,L]     1024
  float* out_te  = out + 1049600;        // [B,L,D]   1048576
  float* out_gab = out + 2098176;        // [B,H,L,L] 16777216
  float* out_x4  = out + 18875392;       // [B,L,D]   1048576
  float* out_xp  = out + 19923968;       // [L,L,B,P] 33554432

  char* ws = (char*)d_ws;
  __bf16* tw1t    = (__bf16*)(ws + 0);          // 2 MB
  __bf16* tw2t    = (__bf16*)(ws + 2097152);    // 2 MB
  __bf16* bw1t    = (__bf16*)(ws + 4194304);    // 128 KB
  __bf16* bw2t    = (__bf16*)(ws + 4325376);    // 32 KB
  __bf16* te      = (__bf16*)(ws + 4358144);    // 2 MB
  __bf16* ht      = (__bf16*)(ws + 6455296);    // 2 MB
  float*  mean_ws = (float*) (ws + 8552448);    // 4 MB
  float*  xpi     = (float*) (ws + 12746752);   // 256 KB

  prep_weights<<<8512, 256, 0, stream>>>(t_w1, t_w2, bp_w1, bp_w2, tw1t, tw2t, bw1t, bw2t);
  embed_kernel<<<1024, 256, 0, stream>>>(token_id, chain_ids, is_periodic, node_attr,
                                         embed_w, atom_feat_w, chain_w, out_x0, out_x4, out_pad);
  time_sinus<<<4096, 256, 0, stream>>>(time_step, te);
  time_gemm<<<512, 256, 0, stream>>>(te, tw1t, t_b1, ht, (float*)nullptr, (const int*)nullptr, 0);
  time_gemm<<<512, 256, 0, stream>>>(ht, tw2t, t_b2, (__bf16*)nullptr, out_te, clean_mask, 1);
  bond_bias<<<8192, 128, 0, stream>>>(token_id, is_periodic, adj, node_type_edge, bond_w,
                                      bw1t, bw2t, bp_ln_g, bp_ln_b, out_gab, mean_ws);
  pair_proj<<<1024, 256, 0, stream>>>(out_x0, pp_ln_g, pp_ln_b, pp_w1, pp_w2, xpi);
  xpair_kernel<<<32768, 256, 0, stream>>>(xpi, mean_ws, out_xp);
}
